// CrossTransFormer_86526411145621
// MI455X (gfx1250) — matrix operand order repaired and checked by reading, not compiled, not run on hardware
//
#include <hip/hip_runtime.h>
#include <hip/hip_bf16.h>

// ---------------------------------------------------------------------------
// MI455X (gfx1250) CrossTransFormer forward.
//   B=8, C=773, P=4096.  Compute-bound (AI ~1000 FLOP/B >> ~200 FLOP/B machine
//   balance) -> bf16 WMMA 16x16x32 with f32 accumulate, flash-softmax so the
//   8 x 4096 x 4096 score matrix never hits HBM.
// ---------------------------------------------------------------------------

static constexpr int NB   = 8;     // batch
static constexpr int NC   = 773;   // channels
static constexpr int NP   = 4096;  // points
static constexpr int CPAD = 800;   // channel pad: 25 * 32 (WMMA k-dim chunks)
static constexpr int NPAD = 896;   // out-channel pad: 8 waves * 7 tiles * 16

typedef __attribute__((ext_vector_type(16))) __bf16 bf16x16;
typedef __attribute__((ext_vector_type(8)))  float  f32x8;
typedef __attribute__((ext_vector_type(4)))  int    i32x4;
typedef __attribute__((ext_vector_type(4)))  float  f32x4;

struct B128x2 { i32x4 lo; i32x4 hi; };

__device__ __forceinline__ bf16x16 frag2(const void* p0, const void* p1) {
  B128x2 t;
  t.lo = *(const i32x4*)p0;
  t.hi = *(const i32x4*)p1;
  return __builtin_bit_cast(bf16x16, t);
}

// A fragment, 16x32 bf16, row-major source (ld in elements).
// lanes 0-15 -> (M=lane, K=0..15); lanes 16-31 -> (M=lane-16, K=16..31)
__device__ __forceinline__ bf16x16 load_a16x32(const __bf16* base, int ld, int lane) {
  const int lr = lane & 15, hf = lane >> 4;
  const __bf16* r = base + (size_t)lr * ld + hf * 16;
  return frag2(r, r + 8);
}

// B fragment, 32x16 bf16, read from B^T stored row-major (N rows x K cols, ld in elements).
// lanes 0-15 -> (N=lane, K=0..15); lanes 16-31 -> (N=lane-16, K=16..31)
__device__ __forceinline__ bf16x16 load_b32x16(const __bf16* baseT, int ld, int lane) {
  const int lr = lane & 15, hf = lane >> 4;
  const __bf16* r = baseT + (size_t)lr * ld + hf * 16;
  return frag2(r, r + 8);
}

// B fragment built on the fly from an f32 row (Q path), with row-validity guard.
__device__ __forceinline__ bf16x16 cvt_b32x16_f32(const float* rowT, bool valid) {
  union U { bf16x16 v; __bf16 e[16]; } u;
  if (valid) {
#pragma unroll
    for (int q = 0; q < 4; ++q) {
      f32x4 f = *(const f32x4*)(rowT + 4 * q);
#pragma unroll
      for (int t = 0; t < 4; ++t) u.e[4 * q + t] = (__bf16)f[t];
    }
  } else {
#pragma unroll
    for (int t = 0; t < 16; ++t) u.e[t] = (__bf16)0.f;
  }
  return u.v;
}

__device__ __forceinline__ f32x8 zero8() {
  f32x8 z = {0.f, 0.f, 0.f, 0.f, 0.f, 0.f, 0.f, 0.f};
  return z;
}

__device__ __forceinline__ f32x8 wmma_bf16(bf16x16 a, bf16x16 b, f32x8 c) {
  return __builtin_amdgcn_wmma_f32_16x16x32_bf16(false, a, false, b, (short)0, c,
                                                 false, false);
}

// ---------------------------------------------------------------------------
// Kernel 1: pad + cast weight matrix (773x773 f32) -> (NPAD x CPAD) bf16
// ---------------------------------------------------------------------------
__global__ __launch_bounds__(256) void k_prep_w(const float* __restrict__ W,
                                                __bf16* __restrict__ Wb) {
  int idx = blockIdx.x * 256 + threadIdx.x;
  if (idx >= NPAD * CPAD) return;
  int o = idx / CPAD, c = idx - o * CPAD;
  float v = (o < NC && c < NC) ? W[o * NC + c] : 0.f;
  Wb[idx] = (__bf16)v;
}

// ---------------------------------------------------------------------------
// Kernel 2: projection  Yt[b][p][o] = bf16( sum_c W[o][c] * X[b][c][p] + bias[o] )
//   X  : (B, C, P) f32       Wb : (NPAD x CPAD) bf16 (zero padded)
//   Yt : (B, P, CPAD) bf16   (cols 773..799 forced to 0)
//   One workgroup = 16-wide p-tile; 8 waves x 7 o-tiles = 896 o span.
// ---------------------------------------------------------------------------
__global__ __launch_bounds__(256) void k_proj(const float* __restrict__ X,
                                              const __bf16* __restrict__ Wb,
                                              const float* __restrict__ bias,
                                              __bf16* __restrict__ Yt) {
  __shared__ __align__(16) __bf16 Xt[16][40];  // 16 p x 32 c, padded rows (80B)

  const int b   = blockIdx.y;
  const int p0  = blockIdx.x * 16;
  const int tid = threadIdx.x;
  const int wid = tid >> 5, lane = tid & 31, lr = lane & 15, hf = lane >> 4;

  const float* Xb = X + (size_t)b * NC * NP;

  f32x8 acc[7];
#pragma unroll
  for (int nt = 0; nt < 7; ++nt) acc[nt] = zero8();

  const int cidx = tid >> 4;  // 0..15
  const int pp   = tid & 15;  // 0..15

  for (int cc = 0; cc < CPAD / 32; ++cc) {
    const int cb = cc * 32;
    // Stage X^T tile (transpose + cast): coalesced f32 reads along p.
#pragma unroll
    for (int k = 0; k < 2; ++k) {
      int c = cb + cidx + k * 16;
      float v = (c < NC) ? Xb[(size_t)c * NP + p0 + pp] : 0.f;
      Xt[pp][cidx + k * 16] = (__bf16)v;
    }
    __syncthreads();

    bf16x16 a = load_a16x32(&Xt[0][0], 40, lane);
#pragma unroll
    for (int nt = 0; nt < 7; ++nt) {
      const int o0 = (wid * 7 + nt) * 16;
      bf16x16 bm = load_b32x16(Wb + (size_t)o0 * CPAD + cb, CPAD, lane);
      acc[nt] = wmma_bf16(a, bm, acc[nt]);
    }
    __syncthreads();
  }

  // Epilogue: + bias, cast, store (M = p-row, N = o-col per D-layout).
  __bf16* Yb = Yt + ((size_t)b * NP + p0) * CPAD;
#pragma unroll
  for (int nt = 0; nt < 7; ++nt) {
    const int o = (wid * 7 + nt) * 16 + lr;
    if (o >= CPAD) continue;
    const float bv = (o < NC) ? bias[o] : 0.f;
#pragma unroll
    for (int r = 0; r < 8; ++r) {
      const int prow = r + hf * 8;
      float v = acc[nt][r] + bv;
      Yb[(size_t)prow * CPAD + o] = (__bf16)((o < NC) ? v : 0.f);
    }
  }
}

// ---------------------------------------------------------------------------
// Kernel 3: flash attention.
//   S[j,i] = (1/sqrt(C)) * sum_c Vt[j][c] * Kt[i][c]      (= H[i,j])
//   softmax over i (streamed, online max/sum)
//   O^T[j,k] = sum_i P[j,i] * Xk[b][k][i]  ->  out[b][k][j]
//   Workgroup: j-tile of 16; i in chunks of 128 (one 16x16 tile per wave);
//   each wave accumulates 7 output N-tiles (k span 896, masked to 773).
// ---------------------------------------------------------------------------
__global__ __launch_bounds__(256) void k_attn(const __bf16* __restrict__ Kt,
                                              const __bf16* __restrict__ Vt,
                                              const float* __restrict__ Xk,
                                              float* __restrict__ out) {
  __shared__ __align__(16) __bf16 Smat[16][136];  // 16 j x 128 i (+pad)
  __shared__ float red[8][16];
  __shared__ float mrow[16], lrow[16], frow[16];
  __shared__ __align__(16) float Ot[NPAD][16];    // 56 KB transpose staging

  const int b   = blockIdx.y;
  const int j0  = blockIdx.x * 16;
  const int tid = threadIdx.x;
  const int wid = tid >> 5, lane = tid & 31, lr = lane & 15, hf = lane >> 4;

  if (tid < 16) { mrow[tid] = -1e30f; lrow[tid] = 0.f; }
  __syncthreads();

  const __bf16* Ktb = Kt + (size_t)b * NP * CPAD;
  const __bf16* Vtb = Vt + (size_t)b * NP * CPAD;
  const float*  Qb  = Xk + (size_t)b * NC * NP;

  f32x8 oacc[7];
#pragma unroll
  for (int nt = 0; nt < 7; ++nt) oacc[nt] = zero8();

  const float sscale = 0.03596757f;  // 1/sqrt(773)

  for (int ic = 0; ic < NP / 128; ++ic) {
    const int ibase = ic * 128;
    const int i0    = ibase + wid * 16;

    // ---- scores: S(16x16) = Vt(j,c) x Kt(i,c)^T over c = 0..799 ----
    f32x8 s = zero8();
    for (int cc = 0; cc < CPAD / 32; ++cc) {
      const int cb = cc * 32;
      bf16x16 a  = load_a16x32(Vtb + (size_t)j0 * CPAD + cb, CPAD, lane);
      bf16x16 bm = load_b32x16(Ktb + (size_t)i0 * CPAD + cb, CPAD, lane);
      s = wmma_bf16(a, bm, s);
    }
    if (ic + 1 < NP / 128)  // hint next K i-chunk toward the caches
      __builtin_prefetch((const void*)(Ktb + (size_t)(ibase + 128 + wid * 16 + lr) * CPAD), 0, 1);

    // ---- per-row max within this wave's 16 i's (lanes of a half) ----
#pragma unroll
    for (int r = 0; r < 8; ++r) {
      float v = s[r] * sscale;
      s[r] = v;
      v = fmaxf(v, __shfl_xor(v, 1));
      v = fmaxf(v, __shfl_xor(v, 2));
      v = fmaxf(v, __shfl_xor(v, 4));
      v = fmaxf(v, __shfl_xor(v, 8));
      if (lr == 0) red[wid][r + hf * 8] = v;
    }
    __syncthreads();

    if (tid < 16) {  // combine 8 waves, online-softmax state update
      float mc = red[0][tid];
#pragma unroll
      for (int ww = 1; ww < 8; ++ww) mc = fmaxf(mc, red[ww][tid]);
      float mo = mrow[tid];
      float mn = fmaxf(mo, mc);
      mrow[tid] = mn;
      frow[tid] = __expf(mo - mn);
    }
    __syncthreads();

    // ---- P = exp(S - m); stash bf16 tile in LDS; per-wave row sums ----
#pragma unroll
    for (int r = 0; r < 8; ++r) {
      const int row = r + hf * 8;
      float p = __expf(s[r] - mrow[row]);
      Smat[row][wid * 16 + lr] = (__bf16)p;
      float sv = p;
      sv += __shfl_xor(sv, 1);
      sv += __shfl_xor(sv, 2);
      sv += __shfl_xor(sv, 4);
      sv += __shfl_xor(sv, 8);
      if (lr == 0) red[wid][row] = sv;
    }
    __syncthreads();

    if (tid < 16) {
      float sc = 0.f;
#pragma unroll
      for (int ww = 0; ww < 8; ++ww) sc += red[ww][tid];
      lrow[tid] = lrow[tid] * frow[tid] + sc;
    }

    // ---- rescale accumulators, then O += P(16x128) x Q^T(128x16-tiles) ----
    float fr[8];
#pragma unroll
    for (int r = 0; r < 8; ++r) fr[r] = frow[r + hf * 8];
#pragma unroll
    for (int nt = 0; nt < 7; ++nt)
#pragma unroll
      for (int r = 0; r < 8; ++r) oacc[nt][r] *= fr[r];

#pragma unroll
    for (int sc2 = 0; sc2 < 4; ++sc2) {
      bf16x16 a = load_a16x32(&Smat[0][0] + sc2 * 32, 136, lane);  // ds_load_b128
#pragma unroll
      for (int nt = 0; nt < 7; ++nt) {
        const int k0  = (wid * 7 + nt) * 16;
        const int kr  = k0 + lr;
        const float* qr = Qb + (size_t)kr * NP + ibase + sc2 * 32 + hf * 16;
        bf16x16 bm = cvt_b32x16_f32(qr, kr < NC);
        oacc[nt] = wmma_bf16(a, bm, oacc[nt]);
      }
    }
    __syncthreads();  // protect red/Smat/frow across chunk boundary
  }

  // ---- finalize: /= l, transpose through LDS, coalesced f32 writeback ----
  float inv[8];
#pragma unroll
  for (int r = 0; r < 8; ++r) inv[r] = 1.0f / lrow[r + hf * 8];
#pragma unroll
  for (int nt = 0; nt < 7; ++nt) {
    const int kk = (wid * 7 + nt) * 16 + lr;
#pragma unroll
    for (int r = 0; r < 8; ++r) Ot[kk][r + hf * 8] = oacc[nt][r] * inv[r];
  }
  __syncthreads();

  float* ob = out + (size_t)b * NC * NP + j0;
  for (int k = tid; k < NC; k += 256) {
    float* dst = ob + (size_t)k * NP;
#pragma unroll
    for (int q = 0; q < 4; ++q)
      *(f32x4*)(dst + 4 * q) = *(const f32x4*)&Ot[k][4 * q];
  }
}

// ---------------------------------------------------------------------------
// Host launcher.  d_in: [Xq, Xk, Wk, bk, Wv, bv]  (all f32)
// Workspace layout (bf16): Kt (B,P,CPAD) | Vt (B,P,CPAD) | Wkb | Wvb  ~106 MB
// ---------------------------------------------------------------------------
extern "C" void kernel_launch(void* const* d_in, const int* in_sizes, int n_in,
                              void* d_out, int out_size, void* d_ws, size_t ws_size,
                              hipStream_t stream) {
  (void)in_sizes; (void)n_in; (void)out_size; (void)ws_size;

  const float* Xq = (const float*)d_in[0];
  const float* Xk = (const float*)d_in[1];
  const float* Wk = (const float*)d_in[2];
  const float* bk = (const float*)d_in[3];
  const float* Wv = (const float*)d_in[4];
  const float* bv = (const float*)d_in[5];
  float* out = (float*)d_out;

  char* ws = (char*)d_ws;
  const size_t ktBytes = (size_t)NB * NP * CPAD * sizeof(__bf16);  // 52,428,800
  const size_t wBytes  = (size_t)NPAD * CPAD * sizeof(__bf16);     //  1,433,600
  __bf16* Kt  = (__bf16*)(ws);
  __bf16* Vt  = (__bf16*)(ws + ktBytes);
  __bf16* Wkb = (__bf16*)(ws + 2 * ktBytes);
  __bf16* Wvb = (__bf16*)(ws + 2 * ktBytes + wBytes);

  const int wElems  = NPAD * CPAD;
  const int wBlocks = (wElems + 255) / 256;
  k_prep_w<<<wBlocks, 256, 0, stream>>>(Wk, Wkb);
  k_prep_w<<<wBlocks, 256, 0, stream>>>(Wv, Wvb);

  dim3 grid(NP / 16, NB);
  k_proj<<<grid, 256, 0, stream>>>(Xk, Wkb, bk, Kt);  // K = Wk*Xk + bk
  k_proj<<<grid, 256, 0, stream>>>(Xq, Wvb, bv, Vt);  // V = Wv*Xq + bv

  k_attn<<<grid, 256, 0, stream>>>(Kt, Vt, Xk, out);
}
